// APPNPTransformerBlock_52037823758877
// MI455X (gfx1250) — matrix operand order repaired and checked by reading, not compiled, not run on hardware
//
#include <hip/hip_runtime.h>
#include <hip/hip_bf16.h>
#include <math.h>

#define N_NODES 4096
#define F_IN    128
#define NH      4
#define HD      32
#define NCLS    64
#define NEDGE   131072
#define KSTEPS  10
#define ALPHA   0.1f

typedef __attribute__((ext_vector_type(16))) _Float16 v16h;
typedef __attribute__((ext_vector_type(8)))  float    v8f;

// ---------------- WMMA helpers (A and B both use contiguous k order) ----------------

// A-matrix 16x32 f16: lane L holds row (L&15); element j -> K = kb + j, kb=(L>>4)*16
__device__ __forceinline__ v16h load_a16(const _Float16* base, int ld, int k0, int lane) {
    int r  = lane & 15;
    int kb = (lane >> 4) << 4;
    const _Float16* p = base + (size_t)r * ld + k0 + kb;
    v16h a;
#pragma unroll
    for (int j = 0; j < 16; ++j) a[j] = p[j];
    return a;
}

// B-matrix 32x16 f16 (B[k][n] = SrcRow[n][k] for a row-major source whose rows are B columns):
// lane L holds col (L&15); element j -> k = k0 + (L>>4)*16 + j (contiguous 32B per lane)
__device__ __forceinline__ v16h load_b16(const _Float16* base, int ld, int k0, int lane) {
    const _Float16* p = base + (size_t)(lane & 15) * ld + k0 + ((lane >> 4) << 4);
    v16h b;
#pragma unroll
    for (int j = 0; j < 16; ++j) b[j] = p[j];
    return b;
}

__device__ __forceinline__ v8f wmma16(v16h a, v16h b, v8f c) {
    return __builtin_amdgcn_wmma_f32_16x16x32_f16(false, a, false, b, (short)0, c, false, false);
}

__device__ __forceinline__ float eluf(float x) { return x > 0.f ? x : (__expf(x) - 1.f); }

// ---------------- Kernel 1: fp32 -> f16 conversions + deg init (self loops) ----------------
__global__ void k_init(const float* __restrict__ data, const float* __restrict__ Wqkv,
                       const float* __restrict__ W1, const float* __restrict__ W2,
                       _Float16* Xh, _Float16* Wqkvh, _Float16* W1h, _Float16* W2h, float* deg) {
    int i = blockIdx.x * blockDim.x + threadIdx.x;
    if (i < N_NODES * F_IN) { Xh[i] = (_Float16)data[i]; return; }
    i -= N_NODES * F_IN;
    if (i < 384 * F_IN)     { Wqkvh[i] = (_Float16)Wqkv[i]; return; }
    i -= 384 * F_IN;
    if (i < 128 * F_IN)     { W1h[i] = (_Float16)W1[i]; return; }
    i -= 128 * F_IN;
    if (i < NCLS * 256)     { W2h[i] = (_Float16)W2[i]; return; }
    i -= NCLS * 256;
    if (i < N_NODES)        { deg[i] = 1.0f; }   // self loop
}

// ---------------- Kernel 2: fused QKV (cols 0..383) + g1 (cols 384..511) GEMM ----------------
// Y[4096,512] = X[4096,128] @ [Wqkv;W1]^T + bias ; one wave per 16x16 tile, 4 K-steps
__global__ __launch_bounds__(128) void k_gemm_qkv(const _Float16* __restrict__ Xh,
                                                  const _Float16* __restrict__ Wqkvh,
                                                  const _Float16* __restrict__ W1h,
                                                  const float* __restrict__ bqkv,
                                                  const float* __restrict__ b1,
                                                  _Float16* QKVh, _Float16* Gh) {
    int lane = threadIdx.x & 31;
    int wave = blockIdx.x * 4 + (threadIdx.x >> 5);   // 0..8191
    int mt = wave >> 5, nt = wave & 31;
    int m0 = mt * 16, col0 = nt * 16;                 // col0 is wave-uniform
    const bool qkv_part = (col0 < 384);
    const _Float16* Bbase = qkv_part ? (Wqkvh + (size_t)col0 * F_IN)
                                     : (W1h + (size_t)(col0 - 384) * F_IN);
    const _Float16* Abase = Xh + (size_t)m0 * F_IN;
    v8f acc = {};
#pragma unroll
    for (int ks = 0; ks < 4; ++ks) {
        v16h a = load_a16(Abase, F_IN, ks * 32, lane);
        v16h b = load_b16(Bbase, F_IN, ks * 32, lane);
        acc = wmma16(a, b, acc);
    }
    int col = col0 + (lane & 15);
    int rbase = m0 + ((lane >> 4) << 3);
    if (qkv_part) {                                   // uniform branch
        float bias = bqkv[col];
#pragma unroll
        for (int r = 0; r < 8; ++r)
            QKVh[(size_t)(rbase + r) * 384 + col] = (_Float16)(acc[r] + bias);
    } else {
        float bias = b1[col - 384];
#pragma unroll
        for (int r = 0; r < 8; ++r)
            Gh[(size_t)(rbase + r) * 256 + (col - 384)] = (_Float16)eluf(acc[r] + bias);
    }
}

// ---------------- Kernel 3: flash attention, 1 wave / (head, 16-row Q tile) ----------------
// QKVh row layout per node: head h -> q at h*96, k at h*96+32, v at h*96+64
// Softmax runs in the log2 domain (v_exp_f32 is base-2): p = 2^(s*scale*log2e - m2)
__global__ __launch_bounds__(32) void k_attn(const _Float16* __restrict__ QKV, _Float16* Gh) {
    __shared__ _Float16 Pld[16 * 32];   // P tile (16 rows x 32 keys), restaged C->A layout
    __shared__ _Float16 Vld[32 * 32];   // V tile (32 keys x 32 dims), row-major
    int lane = threadIdx.x;
    int head = blockIdx.x & 3;
    int qt   = blockIdx.x >> 2;
    const float scale2 = 0.17677669529663687f * 1.4426950408889634f;  // (1/sqrt(32))*log2(e)

    const _Float16* qbase = QKV + (size_t)(qt * 16) * 384 + head * 96;
    v16h aQ = load_a16(qbase, 384, 0, lane);

    float m[8], l[8];                    // m: row max (log2 domain); l: per-lane PARTIAL sums
    v8f o0 = {}, o1 = {};
#pragma unroll
    for (int r = 0; r < 8; ++r) { m[r] = -1e30f; l[r] = 0.f; }

    int colB  = lane & 15;
    int koffB = head * 96 + 32 + ((lane >> 4) << 4);

    for (int j0 = 0; j0 < N_NODES; j0 += 32) {
        // K tiles in B layout: contiguous 16-half load per lane
        v16h bK0, bK1;
        const _Float16* p0 = QKV + (size_t)(j0 + colB) * 384 + koffB;
        const _Float16* p1 = QKV + (size_t)(j0 + 16 + colB) * 384 + koffB;
#pragma unroll
        for (int j = 0; j < 16; ++j) { bK0[j] = p0[j]; bK1[j] = p1[j]; }

        v8f zc = {};
        v8f s0 = wmma16(aQ, bK0, zc);
        v8f s1 = wmma16(aQ, bK1, zc);

        // stage V tile to LDS (lane loads its own key's 32 dims, contiguous)
        {
            const _Float16* pv = QKV + (size_t)(j0 + lane) * 384 + head * 96 + 64;
#pragma unroll
            for (int d = 0; d < 32; ++d) Vld[lane * 32 + d] = pv[d];
        }
        if (j0 + 32 < N_NODES)
            __builtin_prefetch(QKV + (size_t)(j0 + 32 + lane) * 384 + head * 96, 0, 1);

        // online softmax: rows of the C layout live across the 16-lane half groups
#pragma unroll
        for (int r = 0; r < 8; ++r) {
            float a0 = s0[r] * scale2, a1 = s1[r] * scale2;
            float t = fmaxf(a0, a1);
            t = fmaxf(t, __shfl_xor(t, 1, 32));
            t = fmaxf(t, __shfl_xor(t, 2, 32));
            t = fmaxf(t, __shfl_xor(t, 4, 32));
            t = fmaxf(t, __shfl_xor(t, 8, 32));
            float mnew = fmaxf(m[r], t);
            float pv0 = exp2f(a0 - mnew), pv1 = exp2f(a1 - mnew);
            float sf  = exp2f(m[r] - mnew);
            l[r] = l[r] * sf + (pv0 + pv1);   // per-lane partial; reduced once at the end
            m[r] = mnew;
            o0[r] *= sf; o1[r] *= sf;
            s0[r] = pv0; s1[r] = pv1;
        }

        // restage P (C layout -> LDS row-major -> A layout)
        int prow = ((lane >> 4) << 3);
        int pcol = lane & 15;
#pragma unroll
        for (int r = 0; r < 8; ++r) {
            Pld[(prow + r) * 32 + pcol]      = (_Float16)s0[r];
            Pld[(prow + r) * 32 + 16 + pcol] = (_Float16)s1[r];
        }
        __syncthreads();

        v16h aP = load_a16(Pld, 32, 0, lane);
        v16h bV0, bV1;
        {
            int kb16 = ((lane >> 4) << 4);
            int d = lane & 15;
#pragma unroll
            for (int j = 0; j < 16; ++j) {
                bV0[j] = Vld[(kb16 + j) * 32 + d];
                bV1[j] = Vld[(kb16 + j) * 32 + 16 + d];
            }
        }
        o0 = wmma16(aP, bV0, o0);
        o1 = wmma16(aP, bV1, o1);
        __syncthreads();
    }

    // final cross-lane reduction of l (all lanes of a row share the same m history)
#pragma unroll
    for (int r = 0; r < 8; ++r) {
        float s = l[r];
        s += __shfl_xor(s, 1, 32);
        s += __shfl_xor(s, 2, 32);
        s += __shfl_xor(s, 4, 32);
        s += __shfl_xor(s, 8, 32);
        l[r] = s;
    }

    // normalize, ELU, write into concat half of G (cols 128..255)
    int rbase = qt * 16 + ((lane >> 4) << 3);
    int c0 = 128 + head * 32 + (lane & 15);
#pragma unroll
    for (int r = 0; r < 8; ++r) {
        float inv = 1.f / l[r];
        Gh[(size_t)(rbase + r) * 256 + c0]      = (_Float16)eluf(o0[r] * inv);
        Gh[(size_t)(rbase + r) * 256 + c0 + 16] = (_Float16)eluf(o1[r] * inv);
    }
}

// ---------------- Kernel 4: h = elu(G @ W2^T + b2), seeds APPNP state ----------------
__global__ __launch_bounds__(128) void k_gemm_h(const _Float16* __restrict__ Gh,
                                                const _Float16* __restrict__ W2h,
                                                const float* __restrict__ b2,
                                                float* Hb, float* X0) {
    int lane = threadIdx.x & 31;
    int wave = blockIdx.x * 4 + (threadIdx.x >> 5);   // 0..255
    int m0 = wave * 16;
    const _Float16* Abase = Gh + (size_t)m0 * 256;
    v8f acc0 = {}, acc1 = {}, acc2 = {}, acc3 = {};
#pragma unroll
    for (int ks = 0; ks < 8; ++ks) {
        v16h a = load_a16(Abase, 256, ks * 32, lane);
        acc0 = wmma16(a, load_b16(W2h + (size_t)0  * 256, 256, ks * 32, lane), acc0);
        acc1 = wmma16(a, load_b16(W2h + (size_t)16 * 256, 256, ks * 32, lane), acc1);
        acc2 = wmma16(a, load_b16(W2h + (size_t)32 * 256, 256, ks * 32, lane), acc2);
        acc3 = wmma16(a, load_b16(W2h + (size_t)48 * 256, 256, ks * 32, lane), acc3);
    }
    int rbase = m0 + ((lane >> 4) << 3);
    v8f accs[4] = {acc0, acc1, acc2, acc3};
#pragma unroll
    for (int nt = 0; nt < 4; ++nt) {
        int col = nt * 16 + (lane & 15);
        float bias = b2[col];
#pragma unroll
        for (int r = 0; r < 8; ++r) {
            float v = eluf(accs[nt][r] + bias);
            size_t idx = (size_t)(rbase + r) * NCLS + col;
            Hb[idx] = v; X0[idx] = v;
        }
    }
}

// ---------------- APPNP support ----------------
__global__ void k_deg(const int* __restrict__ ei, float* deg) {
    int e = blockIdx.x * blockDim.x + threadIdx.x;
    if (e >= NEDGE) return;
    atomicAdd(&deg[ei[NEDGE + e]], 1.0f);
}

__global__ void k_dinv(const float* __restrict__ deg, float* dinv) {
    int i = blockIdx.x * blockDim.x + threadIdx.x;
    if (i < N_NODES) dinv[i] = rsqrtf(deg[i]);   // deg >= 1 (self loop)
}

// y = alpha*h + (1-alpha)*dinv^2*x   (teleport + self-loop term, full init each round)
__global__ void k_prop_init(const float* __restrict__ h, const float* __restrict__ x,
                            const float* __restrict__ dinv, float* y) {
    int i = blockIdx.x * blockDim.x + threadIdx.x;
    if (i >= N_NODES * NCLS) return;
    float dv = dinv[i >> 6];
    y[i] = ALPHA * h[i] + (1.f - ALPHA) * dv * dv * x[i];
}

// y[col] += (1-alpha)*dinv[row]*dinv[col]*x[row]  per edge per class (coalesced atomics)
__global__ void k_prop_edge(const int* __restrict__ ei, const float* __restrict__ x,
                            const float* __restrict__ dinv, float* y) {
    int idx = blockIdx.x * blockDim.x + threadIdx.x;
    if (idx >= NEDGE * NCLS) return;
    int e = idx >> 6, c = idx & 63;
    int r = ei[e], cl = ei[NEDGE + e];
    float w = (1.f - ALPHA) * dinv[r] * dinv[cl];
    atomicAdd(&y[(size_t)cl * NCLS + c], w * x[(size_t)r * NCLS + c]);
}

// ---------------- log_softmax + raw copy (tuple output) ----------------
__global__ __launch_bounds__(256) void k_lsm(const float* __restrict__ x, float* out) {
    int lane = threadIdx.x & 31;
    int row  = blockIdx.x * 8 + (threadIdx.x >> 5);
    const float* xr = x + (size_t)row * NCLS;
    float v0 = xr[lane], v1 = xr[lane + 32];
    float mx = fmaxf(v0, v1);
    for (int off = 1; off < 32; off <<= 1) mx = fmaxf(mx, __shfl_xor(mx, off, 32));
    float s = __expf(v0 - mx) + __expf(v1 - mx);
    for (int off = 1; off < 32; off <<= 1) s += __shfl_xor(s, off, 32);
    float lse = mx + __logf(s);
    out[(size_t)row * NCLS + lane]      = v0 - lse;
    out[(size_t)row * NCLS + lane + 32] = v1 - lse;
    out[(size_t)N_NODES * NCLS + (size_t)row * NCLS + lane]      = v0;
    out[(size_t)N_NODES * NCLS + (size_t)row * NCLS + lane + 32] = v1;
}

extern "C" void kernel_launch(void* const* d_in, const int* in_sizes, int n_in,
                              void* d_out, int out_size, void* d_ws, size_t ws_size,
                              hipStream_t stream) {
    (void)in_sizes; (void)n_in; (void)out_size; (void)ws_size;
    const float* data = (const float*)d_in[0];
    const int*   ei   = (const int*)d_in[1];     // edge_index [2, E]
    const float* Wqkv = (const float*)d_in[2];
    const float* bqkv = (const float*)d_in[3];
    const float* W1   = (const float*)d_in[4];
    const float* b1   = (const float*)d_in[5];
    const float* W2   = (const float*)d_in[6];
    const float* b2   = (const float*)d_in[7];
    float* out = (float*)d_out;

    char* ws = (char*)d_ws; size_t off = 0;
    auto alloc = [&](size_t bytes) -> char* {
        char* p = ws + off; off += (bytes + 255) & ~(size_t)255; return p;
    };
    _Float16* Xh    = (_Float16*)alloc((size_t)N_NODES * F_IN * 2);
    _Float16* Wqkvh = (_Float16*)alloc((size_t)384 * F_IN * 2);
    _Float16* W1h   = (_Float16*)alloc((size_t)128 * F_IN * 2);
    _Float16* W2h   = (_Float16*)alloc((size_t)NCLS * 256 * 2);
    _Float16* QKVh  = (_Float16*)alloc((size_t)N_NODES * 384 * 2);
    _Float16* Gh    = (_Float16*)alloc((size_t)N_NODES * 256 * 2);
    float*    Hb    = (float*)alloc((size_t)N_NODES * NCLS * 4);
    float*    X0    = (float*)alloc((size_t)N_NODES * NCLS * 4);
    float*    X1    = (float*)alloc((size_t)N_NODES * NCLS * 4);
    float*    deg   = (float*)alloc((size_t)N_NODES * 4);
    float*    dinv  = (float*)alloc((size_t)N_NODES * 4);

    int initN = N_NODES * F_IN + 384 * F_IN + 128 * F_IN + NCLS * 256 + N_NODES;
    k_init<<<(initN + 255) / 256, 256, 0, stream>>>(data, Wqkv, W1, W2, Xh, Wqkvh, W1h, W2h, deg);
    k_gemm_qkv<<<2048, 128, 0, stream>>>(Xh, Wqkvh, W1h, bqkv, b1, QKVh, Gh);
    k_attn<<<(N_NODES / 16) * NH, 32, 0, stream>>>(QKVh, Gh);
    k_gemm_h<<<64, 128, 0, stream>>>(Gh, W2h, b2, Hb, X0);
    k_deg<<<NEDGE / 256, 256, 0, stream>>>(ei, deg);
    k_dinv<<<N_NODES / 256, 256, 0, stream>>>(deg, dinv);

    float* xc = X0; float* xn = X1;
    for (int k = 0; k < KSTEPS; ++k) {
        k_prop_init<<<(N_NODES * NCLS) / 256, 256, 0, stream>>>(Hb, xc, dinv, xn);
        k_prop_edge<<<(NEDGE * NCLS) / 256, 256, 0, stream>>>(ei, xc, dinv, xn);
        float* t = xc; xc = xn; xn = t;
    }
    k_lsm<<<N_NODES / 8, 256, 0, stream>>>(xc, out);
}
